// TransformerEncoderLayerCustom_36043365548561
// MI455X (gfx1250) — matrix operand order repaired and checked by reading, not compiled, not run on hardware
//
#include <hip/hip_runtime.h>

// ---------------------------------------------------------------------------
// CDNA5 (gfx1250) transformer encoder layer: bf16 WMMA GEMMs + flash attention
// + async global->LDS tile staging (ASYNCcnt) with double buffering
// ---------------------------------------------------------------------------

typedef unsigned short u16;
typedef __attribute__((ext_vector_type(16))) __bf16 v16bf;
typedef __attribute__((ext_vector_type(8)))  float  v8f;

#define B_   4
#define S_   2048
#define D_   1024
#define H_   8
#define DH_  128
#define DFF_ 4096
#define M_   (B_ * S_)   // 8192 rows

struct alignas(16) U4 { unsigned int a, b, c, d; };
struct alignas(8)  U2 { unsigned int a, b; };

__device__ __forceinline__ u16 f2bf(float f) {
  unsigned int x = __float_as_uint(f);
  x += 0x7FFFu + ((x >> 16) & 1u);      // round-to-nearest-even
  return (u16)(x >> 16);
}

union FragU { v16bf v; U4 q[2]; };

// load 16 bf16 as two 16-byte chunks (p1 may be p0+8 for contiguous 32B)
__device__ __forceinline__ v16bf load_frag_pair(const u16* p0, const u16* p1) {
  FragU u;
  u.q[0] = *reinterpret_cast<const U4*>(p0);
  u.q[1] = *reinterpret_cast<const U4*>(p1);
  return u.v;
}

// --- CDNA5 async global->LDS (ASYNCcnt) ------------------------------------
__device__ __forceinline__ void async_load_b128(const void* lds_dst, const void* gaddr) {
  unsigned loff = (unsigned)(unsigned long long)lds_dst;   // low 32b = LDS offset
  asm volatile("global_load_async_to_lds_b128 %0, %1, off"
               :: "v"(loff), "v"(gaddr) : "memory");
}
__device__ __forceinline__ void async_wait0() {
  asm volatile("s_wait_asynccnt 0x0" ::: "memory");
}

// ---------------------------------------------------------------------------
// GEMM: C[M][N] = A[M][K](bf16) x Bt[N][K](bf16)^T  (+bias, +relu, +residual)
// block tile 128x128, K-step 32, 256 threads = 8 waves (4 M-slices x 2 N-slices)
// double-buffered LDS; tiles staged with global_load_async_to_lds_b128
// ---------------------------------------------------------------------------
#define LDK 40   // 32 + 8 pad halfwords -> 80B row stride, bank-conflict free

// each thread issues 2 x b128 per matrix tile: 512 chunks of 16B = 8KB tile
__device__ __forceinline__ void issue_tile_async(const u16* gbase, int K, int k0,
                                                 u16* ldsBase, int tid) {
#pragma unroll
  for (int i = 0; i < 2; i++) {
    const int ch = tid + i * 256;       // 0..511
    const int row = ch >> 2;            // 0..127
    const int part = ch & 3;            // 16B chunk within the 64B row
    async_load_b128(&ldsBase[row * LDK + part * 8],
                    gbase + (size_t)row * K + k0 + part * 8);
  }
}

__global__ __launch_bounds__(256)
void gemm_bf16(const u16* __restrict__ A, const u16* __restrict__ Bt,
               int M, int N, int K,
               const float* __restrict__ bias,
               const float* __restrict__ resid,
               float* __restrict__ outF, u16* __restrict__ outB,
               int relu)
{
  __shared__ __align__(16) u16 ldsA[2][128 * LDK];
  __shared__ __align__(16) u16 ldsB[2][128 * LDK];

  const int tid   = threadIdx.x;
  const int lane  = tid & 31;
  const int wid   = tid >> 5;
  const int hlf   = lane >> 4;
  const int l     = lane & 15;
  const int waveM = wid & 3;     // 32-row slice
  const int waveN = wid >> 2;    // 64-col slice
  const int mBlock = blockIdx.y * 128;
  const int nBlock = blockIdx.x * 128;

  const u16* Ab = A  + (size_t)mBlock * K;
  const u16* Bb = Bt + (size_t)nBlock * K;

  v8f acc[2][4];
#pragma unroll
  for (int mi = 0; mi < 2; mi++)
#pragma unroll
    for (int ni = 0; ni < 4; ni++)
#pragma unroll
      for (int r = 0; r < 8; r++) acc[mi][ni][r] = 0.f;

  // prologue: stage tile 0
  issue_tile_async(Ab, K, 0, ldsA[0], tid);
  issue_tile_async(Bb, K, 0, ldsB[0], tid);
  async_wait0();

  const int nK = K >> 5;
  for (int kt = 0; kt < nK; kt++) {
    __syncthreads();   // tile kt in LDS visible; buf (kt+1)&1 free for refill

    if (kt + 1 < nK) { // prefetch next tile while computing this one
      issue_tile_async(Ab, K, (kt + 1) * 32, ldsA[(kt + 1) & 1], tid);
      issue_tile_async(Bb, K, (kt + 1) * 32, ldsB[(kt + 1) & 1], tid);
    }

    const u16* bufA = ldsA[kt & 1];
    const u16* bufB = ldsB[kt & 1];

    v16bf aF[2], bF[4];
#pragma unroll
    for (int mi = 0; mi < 2; mi++) {
      const u16* p0 = &bufA[(waveM * 32 + mi * 16 + l) * LDK + hlf * 8];
      aF[mi] = load_frag_pair(p0, p0 + 16);
    }
#pragma unroll
    for (int ni = 0; ni < 4; ni++) {
      const u16* p = &bufB[(waveN * 64 + ni * 16 + l) * LDK + hlf * 8];
      bF[ni] = load_frag_pair(p, p + 16);
    }
#pragma unroll
    for (int mi = 0; mi < 2; mi++)
#pragma unroll
      for (int ni = 0; ni < 4; ni++)
        acc[mi][ni] = __builtin_amdgcn_wmma_f32_16x16x32_bf16(
            false, aF[mi], false, bF[ni], (short)0, acc[mi][ni], false, false);

    async_wait0();     // prefetched tile fully landed before next barrier
  }

  for (int mi = 0; mi < 2; mi++) {
    for (int ni = 0; ni < 4; ni++) {
      const int col = nBlock + waveN * 64 + ni * 16 + l;
      float bv = bias ? bias[col] : 0.f;
      for (int r = 0; r < 8; r++) {
        const int row = mBlock + waveM * 32 + mi * 16 + hlf * 8 + r;
        float v = acc[mi][ni][r] + bv;
        if (relu)  v = fmaxf(v, 0.f);
        if (resid) v += resid[(size_t)row * N + col];
        if (outF)  outF[(size_t)row * N + col] = v;
        if (outB)  outB[(size_t)row * N + col] = f2bf(v);
      }
    }
  }
}

// ---------------------------------------------------------------------------
// Causal flash attention: one block = (batch*head, 64-query tile), 128 threads
// ---------------------------------------------------------------------------
#define KT 64
#define LDSK_STRIDE 136   // 128 + 8 pad
#define LDSV_STRIDE 72    // 64 + 8 pad
#define LDSP_STRIDE 72

__global__ __launch_bounds__(128)
void attention_causal(const u16* __restrict__ Q, const u16* __restrict__ Kg,
                      const u16* __restrict__ Vg, u16* __restrict__ O)
{
  __shared__ __align__(16) u16 ldsK[KT * LDSK_STRIDE];
  __shared__ __align__(16) u16 ldsV[DH_ * LDSV_STRIDE];
  __shared__ __align__(16) u16 ldsP[4 * 16 * LDSP_STRIDE];

  const int tid  = threadIdx.x;
  const int lane = tid & 31;
  const int w    = tid >> 5;        // wave 0..3 -> 16 query rows each
  const int hlf  = lane >> 4;
  const int l    = lane & 15;

  const int qTile = blockIdx.x;     // 0..31
  const int bh    = blockIdx.y;     // 0..31
  const int b     = bh >> 3;
  const int h     = bh & 7;

  const size_t headOff = (size_t)bh * S_ * DH_;
  const u16* Qh = Q  + headOff;
  const u16* Kh = Kg + headOff;
  const u16* Vh = Vg + headOff;

  const int qBase = qTile * 64;
  const int qRowW = qBase + w * 16;

  // preload Q fragments for 4 dh-steps of 32 (Q is pre-scaled by 1/sqrt(DH))
  v16bf qF[4];
#pragma unroll
  for (int c = 0; c < 4; c++) {
    const u16* p0 = Qh + (size_t)(qRowW + l) * DH_ + c * 32 + hlf * 8;
    qF[c] = load_frag_pair(p0, p0 + 16);
  }

  float mI[8], lI[8];
  v8f oAcc[8];
#pragma unroll
  for (int r = 0; r < 8; r++) { mI[r] = -1e30f; lI[r] = 0.f; }
#pragma unroll
  for (int cc = 0; cc < 8; cc++)
#pragma unroll
    for (int r = 0; r < 8; r++) oAcc[cc][r] = 0.f;

  const int ldRow  = tid >> 1;   // 0..63
  const int ldPart = tid & 1;

  u16* myP = &ldsP[w * 16 * LDSP_STRIDE];

  const int nTiles = qTile + 1;
  for (int kt = 0; kt < nTiles; kt++) {
    __syncthreads();
    // K tile -> ldsK[kv][dh] via async DMA (contiguous rows)
#pragma unroll
    for (int i = 0; i < 8; i++) {
      async_load_b128(&ldsK[ldRow * LDSK_STRIDE + ldPart * 64 + i * 8],
                      Kh + (size_t)(kt * KT + ldRow) * DH_ + ldPart * 64 + i * 8);
    }
    // V tile (transposed) -> ldsV[dh][kv] (must round-trip through VGPRs)
    {
      const U4* g = reinterpret_cast<const U4*>(Vh + (size_t)(kt * KT + ldRow) * DH_ + ldPart * 64);
      union { U4 q[8]; u16 s[64]; } tmp;
#pragma unroll
      for (int i = 0; i < 8; i++) tmp.q[i] = g[i];
#pragma unroll
      for (int j = 0; j < 64; j++)
        ldsV[(ldPart * 64 + j) * LDSV_STRIDE + ldRow] = tmp.s[j];
    }
    async_wait0();
    __syncthreads();

    // scores S = Q * K^T : 4 kv-fragments x 4 dh-steps
    v8f sAcc[4];
#pragma unroll
    for (int ni = 0; ni < 4; ni++)
#pragma unroll
      for (int r = 0; r < 8; r++) sAcc[ni][r] = 0.f;
#pragma unroll
    for (int c = 0; c < 4; c++) {
#pragma unroll
      for (int ni = 0; ni < 4; ni++) {
        const u16* p = &ldsK[(ni * 16 + l) * LDSK_STRIDE + c * 32 + hlf * 8];
        v16bf bK = load_frag_pair(p, p + 16);
        sAcc[ni] = __builtin_amdgcn_wmma_f32_16x16x32_bf16(
            false, qF[c], false, bK, (short)0, sAcc[ni], false, false);
      }
    }

    // causal mask on the diagonal tile
    if (kt == qTile) {
#pragma unroll
      for (int ni = 0; ni < 4; ni++) {
        const int kv = ni * 16 + l;
#pragma unroll
        for (int r = 0; r < 8; r++) {
          const int qr = w * 16 + hlf * 8 + r;
          if (kv > qr) sAcc[ni][r] = -1e30f;
        }
      }
    }

    // online softmax (row = hlf*8 + r within wave; 16-lane reductions)
    float p[4][8];
#pragma unroll
    for (int r = 0; r < 8; r++) {
      float m = sAcc[0][r];
#pragma unroll
      for (int ni = 1; ni < 4; ni++) m = fmaxf(m, sAcc[ni][r]);
      for (int off = 1; off < 16; off <<= 1) m = fmaxf(m, __shfl_xor(m, off, 16));
      const float newM = fmaxf(mI[r], m);
      const float corr = __expf(mI[r] - newM);
      float rs = 0.f;
#pragma unroll
      for (int ni = 0; ni < 4; ni++) {
        float e = __expf(sAcc[ni][r] - newM);
        p[ni][r] = e;
        rs += e;
      }
      for (int off = 1; off < 16; off <<= 1) rs += __shfl_xor(rs, off, 16);
      lI[r] = lI[r] * corr + rs;
      mI[r] = newM;
#pragma unroll
      for (int cc = 0; cc < 8; cc++) oAcc[cc][r] *= corr;
    }

    // stash P (bf16) in per-wave LDS patch: D-layout -> row-major
#pragma unroll
    for (int ni = 0; ni < 4; ni++)
#pragma unroll
      for (int r = 0; r < 8; r++)
        myP[(hlf * 8 + r) * LDSP_STRIDE + ni * 16 + l] = f2bf(p[ni][r]);
    __syncthreads();

    // O += P * V : re-read P as A fragments, V from transposed LDS
    v16bf aP[2];
#pragma unroll
    for (int ks = 0; ks < 2; ks++) {
      const u16* p0 = myP + l * LDSP_STRIDE + ks * 32 + hlf * 8;
      aP[ks] = load_frag_pair(p0, p0 + 16);
    }
#pragma unroll
    for (int cc = 0; cc < 8; cc++) {
#pragma unroll
      for (int ks = 0; ks < 2; ks++) {
        const u16* pv = &ldsV[(cc * 16 + l) * LDSV_STRIDE + ks * 32 + hlf * 8];
        v16bf bV = load_frag_pair(pv, pv + 16);
        oAcc[cc] = __builtin_amdgcn_wmma_f32_16x16x32_bf16(
            false, aP[ks], false, bV, (short)0, oAcc[cc], false, false);
      }
    }
  }

  // finalize: O / l, scatter back to [B*S][D] (heads concatenated), bf16
  for (int r = 0; r < 8; r++) {
    const float inv = 1.f / lI[r];
    const size_t row = (size_t)b * S_ + qBase + w * 16 + hlf * 8 + r;
    for (int cc = 0; cc < 8; cc++) {
      const int col = h * DH_ + cc * 16 + l;
      O[row * D_ + col] = f2bf(oAcc[cc][r] * inv);
    }
  }
}

// ---------------------------------------------------------------------------
// helpers: conversions, qkv split, layernorm
// ---------------------------------------------------------------------------
__global__ __launch_bounds__(256)
void convert_f32_to_bf16(const float* __restrict__ in, u16* __restrict__ out, int n4)
{
  int i = blockIdx.x * 256 + threadIdx.x;
  if (i >= n4) return;
  float4 f = reinterpret_cast<const float4*>(in)[i];
  U2 o;
  o.a = (unsigned)f2bf(f.x) | ((unsigned)f2bf(f.y) << 16);
  o.b = (unsigned)f2bf(f.z) | ((unsigned)f2bf(f.w) << 16);
  reinterpret_cast<U2*>(out)[i] = o;
}

__global__ __launch_bounds__(256)
void transpose_f32_to_bf16(const float* __restrict__ in, u16* __restrict__ out, int R, int C)
{
  size_t i = (size_t)blockIdx.x * 256 + threadIdx.x;
  if (i >= (size_t)R * C) return;
  int c = (int)(i / R);
  int r = (int)(i % R);
  out[i] = f2bf(in[(size_t)r * C + c]);
}

__global__ __launch_bounds__(256)
void qkv_split(const float* __restrict__ qkv, u16* __restrict__ q,
               u16* __restrict__ k, u16* __restrict__ v)
{
  size_t i = (size_t)blockIdx.x * 256 + threadIdx.x;
  const size_t total = (size_t)M_ * 3 * D_;
  if (i >= total) return;
  int m = (int)(i / (3 * D_));
  int n = (int)(i % (3 * D_));
  int which = n >> 10;
  int d  = n & 1023;
  int hh = d >> 7;
  int dh = d & 127;
  int bb = m >> 11;
  int ss = m & 2047;
  size_t dst = (((size_t)bb * H_ + hh) * S_ + ss) * DH_ + dh;
  float val = qkv[i];
  if (which == 0)      q[dst] = f2bf(val * 0.08838834764831845f);  // 1/sqrt(128)
  else if (which == 1) k[dst] = f2bf(val);
  else                 v[dst] = f2bf(val);
}

__global__ __launch_bounds__(256)
void layernorm_rows(const float* __restrict__ in, const float* __restrict__ gamma,
                    const float* __restrict__ beta, float* __restrict__ outF,
                    u16* __restrict__ outB)
{
  __shared__ float red[256];
  const int row = blockIdx.x;
  const int t = threadIdx.x;
  const float* x = in + (size_t)row * D_;
  float4 v = reinterpret_cast<const float4*>(x)[t];
  red[t] = v.x + v.y + v.z + v.w;
  __syncthreads();
  for (int off = 128; off > 0; off >>= 1) {
    if (t < off) red[t] += red[t + off];
    __syncthreads();
  }
  const float mu = red[0] * (1.f / D_);
  __syncthreads();
  const float dx = v.x - mu, dy = v.y - mu, dz = v.z - mu, dw = v.w - mu;
  red[t] = dx * dx + dy * dy + dz * dz + dw * dw;
  __syncthreads();
  for (int off = 128; off > 0; off >>= 1) {
    if (t < off) red[t] += red[t + off];
    __syncthreads();
  }
  const float rstd = rsqrtf(red[0] * (1.f / D_) + 1e-5f);
  float4 g  = reinterpret_cast<const float4*>(gamma)[t];
  float4 be = reinterpret_cast<const float4*>(beta)[t];
  float4 o;
  o.x = dx * rstd * g.x + be.x;
  o.y = dy * rstd * g.y + be.y;
  o.z = dz * rstd * g.z + be.z;
  o.w = dw * rstd * g.w + be.w;
  if (outF) reinterpret_cast<float4*>(outF + (size_t)row * D_)[t] = o;
  if (outB) {
    u16* ob = outB + (size_t)row * D_ + t * 4;
    ob[0] = f2bf(o.x); ob[1] = f2bf(o.y); ob[2] = f2bf(o.z); ob[3] = f2bf(o.w);
  }
}

// ---------------------------------------------------------------------------
// launcher
// ---------------------------------------------------------------------------
extern "C" void kernel_launch(void* const* d_in, const int* in_sizes, int n_in,
                              void* d_out, int out_size, void* d_ws, size_t ws_size,
                              hipStream_t stream)
{
  const float* src   = (const float*)d_in[0];
  const float* w_qkv = (const float*)d_in[1];
  const float* w_out = (const float*)d_in[2];
  const float* w1    = (const float*)d_in[3];
  const float* b1    = (const float*)d_in[4];
  const float* w2    = (const float*)d_in[5];
  const float* b2    = (const float*)d_in[6];
  const float* g1    = (const float*)d_in[7];
  const float* be1   = (const float*)d_in[8];
  const float* g2    = (const float*)d_in[9];
  const float* be2   = (const float*)d_in[10];
  float* out = (float*)d_out;

  char* ws = (char*)d_ws;
  size_t off = 0;
  auto alloc = [&](size_t bytes) -> char* {
    char* p = ws + off;
    off += (bytes + 255) & ~(size_t)255;
    return p;
  };

  u16*  wqkvT = (u16*)alloc((size_t)3 * D_ * D_ * 2);     //  6 MB  [3D][D]
  u16*  woutT = (u16*)alloc((size_t)D_ * D_ * 2);         //  2 MB  [D][D]
  u16*  w1T   = (u16*)alloc((size_t)DFF_ * D_ * 2);       //  8 MB  [DFF][D]
  u16*  w2T   = (u16*)alloc((size_t)D_ * DFF_ * 2);       //  8 MB  [D][DFF]
  u16*  srcBf = (u16*)alloc((size_t)M_ * D_ * 2);         // 16 MB
  char* bigR  = alloc((size_t)M_ * 3 * D_ * 4);           // 96 MB (qkv f32 -> h bf16 + y f32)
  u16*  qbf   = (u16*)alloc((size_t)M_ * D_ * 2);         // 16 MB
  u16*  kbf   = (u16*)alloc((size_t)M_ * D_ * 2);         // 16 MB
  u16*  vbf   = (u16*)alloc((size_t)M_ * D_ * 2);         // 16 MB
  u16*  attnO = (u16*)alloc((size_t)M_ * D_ * 2);         // 16 MB

  float* qkvF = (float*)bigR;                             // qkv projection (f32)
  u16*   hBuf = (u16*)bigR;                               // FFN hidden (bf16), reuses qkvF
  float* yBuf = (float*)(bigR + (size_t)M_ * DFF_ * 2);   // final residual sum, reuses qkvF tail
  float* xBuf = (float*)qbf;                              // x1 / x (f32, 32MB spans qbf+kbf)
  u16*   xbf  = vbf;                                      // x in bf16, reuses vbf

  if (ws_size < off) return;

  // 1) precision conversions / weight transposes
  convert_f32_to_bf16<<<(M_ * D_ / 4 + 255) / 256, 256, 0, stream>>>(src, srcBf, M_ * D_ / 4);
  transpose_f32_to_bf16<<<(3 * D_ * D_ + 255) / 256, 256, 0, stream>>>(w_qkv, wqkvT, D_, 3 * D_);
  transpose_f32_to_bf16<<<(D_ * D_ + 255) / 256, 256, 0, stream>>>(w_out, woutT, D_, D_);
  transpose_f32_to_bf16<<<(D_ * DFF_ + 255) / 256, 256, 0, stream>>>(w1, w1T, D_, DFF_);
  transpose_f32_to_bf16<<<(DFF_ * D_ + 255) / 256, 256, 0, stream>>>(w2, w2T, DFF_, D_);

  // 2) qkv = src @ w_qkv
  gemm_bf16<<<dim3(3 * D_ / 128, M_ / 128), 256, 0, stream>>>(
      srcBf, wqkvT, M_, 3 * D_, D_, nullptr, nullptr, qkvF, nullptr, 0);

  // 3) split to per-head bf16 Q (pre-scaled), K, V
  qkv_split<<<(M_ * 3 * D_ + 255) / 256, 256, 0, stream>>>(qkvF, qbf, kbf, vbf);

  // 4) causal flash attention -> attnO [M][D] bf16
  attention_causal<<<dim3(S_ / 64, B_ * H_), 128, 0, stream>>>(qbf, kbf, vbf, attnO);

  // 5) x1 = attnO @ w_out + src (f32)
  gemm_bf16<<<dim3(D_ / 128, M_ / 128), 256, 0, stream>>>(
      attnO, woutT, M_, D_, D_, nullptr, src, xBuf, nullptr, 0);

  // 6) x = LN(x1) in-place, also bf16 copy
  layernorm_rows<<<M_, 256, 0, stream>>>(xBuf, g1, be1, xBuf, xbf);

  // 7) h = relu(x @ w1 + b1) -> bf16
  gemm_bf16<<<dim3(DFF_ / 128, M_ / 128), 256, 0, stream>>>(
      xbf, w1T, M_, DFF_, D_, b1, nullptr, nullptr, hBuf, 1);

  // 8) y = h @ w2 + b2 + x  (f32)
  gemm_bf16<<<dim3(D_ / 128, M_ / 128), 256, 0, stream>>>(
      hBuf, w2T, M_, D_, DFF_, b2, xBuf, yBuf, nullptr, 0);

  // 9) out = LN(y)
  layernorm_rows<<<M_, 256, 0, stream>>>(yBuf, g2, be2, out, nullptr);
}
